// VQVAE_MODEL_13271448944823
// MI455X (gfx1250) — matrix operand order repaired and checked by reading, not compiled, not run on hardware
//
#include <hip/hip_runtime.h>

typedef __attribute__((ext_vector_type(16))) _Float16 v16h;
typedef __attribute__((ext_vector_type(8)))  _Float16 v8h;
typedef __attribute__((ext_vector_type(8)))  float    v8f;

#define WG 256
#define LATENT 16
#define NUM_EMB 128
#define BATCH 256

// ---------------------------------------------------------------------------
// f32 -> f16 conversion (weights + network input)
// ---------------------------------------------------------------------------
__global__ __launch_bounds__(WG)
void cvt_f32_to_f16(const float* __restrict__ in, _Float16* __restrict__ out, int n) {
    int i = blockIdx.x * WG + threadIdx.x;
    if (i < n) out[i] = (_Float16)in[i];
}

// ---------------------------------------------------------------------------
// Conv / conv-transpose as im2col GEMM on the WMMA pipe, fully templated.
//   A : [M, K]  M = BATCH*Ho*Wo positions, K = KH*KW*Ci (im2col, zero-padded)
//   B : [K, Co] HWIO weights; staged TRANSPOSED in LDS so each lane's
//       fragment is two contiguous 16B chunks (ds_load_b128).
//   D = A x B via V_WMMA_F32_16X16X32_F16; K consumed 64/iter.
// Block = 256 threads = 8 wave32; waves tile (8/NWN) M x NWN N positions;
// each wave owns NTW adjacent 16-wide N tiles (A-fragment reuse).
// ASYNC_A: when im2col is the identity (1x1 conv, K%64==0), stage A with
// global_load_async_to_lds_b128 (ASYNCcnt DMA path) instead of VGPR bounce.
// ---------------------------------------------------------------------------
template<int Hi, int Wi, int Ci, int Ho, int Wo, int Co,
         int KH, int KW, int STRIDE, int PAD, int TRANSPOSE, int RELU,
         int NWN, int NTW, bool OUT32, bool ASYNC_A>
__global__ __launch_bounds__(WG)
void conv_gemm_wmma(const _Float16* __restrict__ in,   // [BATCH,Hi,Wi,Ci] f16
                    const _Float16* __restrict__ wgt,  // [K, Co] f16
                    const float*    __restrict__ bias, // [Co] f32
                    _Float16* __restrict__ out16,      // [M, Co] f16
                    float*    __restrict__ out32)      // [M, Co] f32
{
    constexpr int K  = KH * KW * Ci;
    constexpr int M  = BATCH * Ho * Wo;
    constexpr int MT = (8 / NWN) * 16;        // block M tile (64 or 128)
    constexpr int NT = NWN * NTW * 16;        // block N tile (16..64)
    constexpr int KC = 64;                    // K per LDS stage
    constexpr int HW = Ho * Wo;
    constexpr int LSTR = 72;                  // LDS row stride (144B, 16B-aligned)

    __shared__ __align__(16) _Float16 ldsA[128][LSTR];   // [MT rows][KC]
    __shared__ __align__(16) _Float16 ldsBT[64][LSTR];   // [NT cols][KC] (transposed)

    const int lane = threadIdx.x & 31;
    const int wave = threadIdx.x >> 5;
    const int wm   = wave / NWN;
    const int wn   = wave % NWN;

    const int mBlock = blockIdx.x * MT;
    const int nBlock = blockIdx.y * NT;

    v8f acc[NTW];
#pragma unroll
    for (int t = 0; t < NTW; ++t) acc[t] = (v8f){};

    const int hsel = lane >> 4;               // 0: lanes 0-15, 1: lanes 16-31
    const v8h* rowA = (const v8h*)&ldsA[wm * 16 + (lane & 15)][0];
    const v8h* rowB[NTW];
#pragma unroll
    for (int t = 0; t < NTW; ++t)
        rowB[t] = (const v8h*)&ldsBT[(wn * NTW + t) * 16 + (lane & 15)][0];

    for (int k0 = 0; k0 < K; k0 += KC) {
        // prefetch next weight chunk while we work on this one
        if (k0 + KC < K)
            __builtin_prefetch(wgt + (size_t)(k0 + KC) * Co, 0, 1);

        // ---- stage A tile ----
        if (ASYNC_A) {
            // im2col == identity: A rows are contiguous 128B in memory.
            // DMA memory -> LDS, 16B per lane, tracked by ASYNCcnt.
#pragma unroll
            for (int s = threadIdx.x; s < MT * (KC / 8); s += WG) {
                int r   = s >> 3;          // row
                int sub = s & 7;           // 16B segment within the 128B row chunk
                unsigned ldsoff = (unsigned)(unsigned long long)&ldsA[r][sub * 8];
                const _Float16* g = in + (size_t)(mBlock + r) * K + k0 + sub * 8;
                asm volatile("global_load_async_to_lds_b128 %0, %1, off"
                             :: "v"(ldsoff), "v"(g) : "memory");
            }
            asm volatile("s_wait_asynccnt 0x0" ::: "memory");
        } else {
            // general im2col with zero padding
            for (int e = threadIdx.x; e < MT * KC; e += WG) {
                int r  = e >> 6;
                int kk = e & (KC - 1);
                int k  = k0 + kk;
                int m  = mBlock + r;
                _Float16 v = (_Float16)0.0f;
                if (k < K) {
                    int b  = m / HW;
                    int rr = m - b * HW;
                    int oy = rr / Wo;
                    int ox = rr - oy * Wo;
                    int ci = k % Ci;
                    int t  = k / Ci;
                    int ky = t / KW;
                    int kx = t - ky * KW;
                    int iy, ix;
                    bool ok;
                    if (!TRANSPOSE) {
                        iy = oy * STRIDE + ky - PAD;
                        ix = ox * STRIDE + kx - PAD;
                        ok = (iy >= 0) && (iy < Hi) && (ix >= 0) && (ix < Wi);
                    } else {
                        int ty = oy + PAD - ky;
                        int tx = ox + PAD - kx;
                        ok = (ty >= 0) && (tx >= 0) &&
                             ((ty % STRIDE) == 0) && ((tx % STRIDE) == 0);
                        iy = ty / STRIDE;
                        ix = tx / STRIDE;
                        ok = ok && (iy < Hi) && (ix < Wi);
                    }
                    if (ok) v = in[(size_t)((b * Hi + iy) * Wi + ix) * Ci + ci];
                }
                ldsA[r][kk] = v;
            }
        }
        // ---- stage B tile, transposed: ldsBT[n][k] ----
        for (int e = threadIdx.x; e < KC * NT; e += WG) {
            int n  = e & (NT - 1);    // NT is a power of two
            int kk = e >> __builtin_ctz(NT);
            int k  = k0 + kk;
            int nG = nBlock + n;
            _Float16 v = (_Float16)0.0f;
            if (k < K && ((Co % NT == 0) || (nG < Co))) v = wgt[(size_t)k * Co + nG];
            ldsBT[n][kk] = v;
        }
        __syncthreads();

        // ---- 2 x NTW WMMAs over the 64-wide K chunk (A fragment reused) ----
#pragma unroll
        for (int c = 0; c < 2; ++c) {
            // A 16x32: lanes 0-15 K={0..7,16..23}, lanes 16-31 K={8..15,24..31}
            v8h a0 = rowA[c * 4 + hsel];
            v8h a1 = rowA[c * 4 + 2 + hsel];
            v16h av = __builtin_shufflevector(a0, a1, 0,1,2,3,4,5,6,7,8,9,10,11,12,13,14,15);
#pragma unroll
            for (int t = 0; t < NTW; ++t) {
                // B 32x16: lane%16 = N col; lanes 0-15 K={0..7,16..23}, lanes 16-31 K={8..15,24..31}
                v8h b0 = rowB[t][c * 4 + hsel];
                v8h b1 = rowB[t][c * 4 + 2 + hsel];
                v16h bv = __builtin_shufflevector(b0, b1, 0,1,2,3,4,5,6,7,8,9,10,11,12,13,14,15);
                acc[t] = __builtin_amdgcn_wmma_f32_16x16x32_f16(
                    false, av, false, bv, (short)0, acc[t], false, false);
            }
        }
        __syncthreads();
    }

    // ---- epilogue: bias + relu (guards constant-folded where exact) ----
    // C/D layout: lane%16 = N; VGPR r -> M = r (lane<16) or 8+r (lane>=16)
#pragma unroll
    for (int t = 0; t < NTW; ++t) {
        int nG = nBlock + (wn * NTW + t) * 16 + (lane & 15);
        if ((Co % 16 == 0) || (nG < Co)) {
            float bv = bias[nG];
#pragma unroll
            for (int r = 0; r < 8; ++r) {
                int mLocal = wm * 16 + ((lane < 16) ? r : (8 + r));
                int mG = mBlock + mLocal;      // M % MT == 0 for all launches
                float v = acc[t][r] + bv;
                if (RELU) v = fmaxf(v, 0.0f);
                size_t idx = (size_t)mG * Co + nG;
                if (OUT32) out32[idx] = v;
                else       out16[idx] = (_Float16)v;
            }
        }
    }
}

// ---------------------------------------------------------------------------
// Vector quantization: argmin over 128 codes (dim 16); forward output = code.
// ---------------------------------------------------------------------------
__global__ __launch_bounds__(WG)
void vq_kernel(const _Float16* __restrict__ z,   // [N, 16]
               const float*    __restrict__ emb, // [16, 128]
               _Float16* __restrict__ zq,        // [N, 16]
               int N)
{
    __shared__ float e[LATENT][NUM_EMB];
    for (int i = threadIdx.x; i < LATENT * NUM_EMB; i += WG)
        e[i / NUM_EMB][i % NUM_EMB] = emb[i];
    __syncthreads();

    int m = blockIdx.x * WG + threadIdx.x;
    if (m >= N) return;

    float zz[LATENT];
#pragma unroll
    for (int d = 0; d < LATENT; ++d) zz[d] = (float)z[(size_t)m * LATENT + d];

    int best = 0;
    float bestd = 3.4e38f;
    for (int c = 0; c < NUM_EMB; ++c) {
        float s = 0.0f;
#pragma unroll
        for (int d = 0; d < LATENT; ++d) {
            float df = zz[d] - e[d][c];
            s += df * df;
        }
        if (s < bestd) { bestd = s; best = c; }
    }
#pragma unroll
    for (int d = 0; d < LATENT; ++d)
        zq[(size_t)m * LATENT + d] = (_Float16)e[d][best];
}

// ---------------------------------------------------------------------------
// Host-side orchestration
// ---------------------------------------------------------------------------
template<int Hi, int Wi, int Ci, int Ho, int Wo, int Co,
         int KH, int KW, int STRIDE, int PAD, int TRANSPOSE, int RELU,
         int NWN, int NTW, bool OUT32, bool ASYNC_A>
static void launch_conv(const _Float16* in, const _Float16* w, const float* bias,
                        _Float16* o16, float* o32, hipStream_t stream) {
    constexpr int M  = BATCH * Ho * Wo;
    constexpr int MT = (8 / NWN) * 16;
    constexpr int NT = NWN * NTW * 16;
    dim3 grid((unsigned)((M + MT - 1) / MT), (unsigned)((Co + NT - 1) / NT));
    conv_gemm_wmma<Hi, Wi, Ci, Ho, Wo, Co, KH, KW, STRIDE, PAD, TRANSPOSE, RELU,
                   NWN, NTW, OUT32, ASYNC_A>
        <<<grid, dim3(WG), 0, stream>>>(in, w, bias, o16, o32);
}

extern "C" void kernel_launch(void* const* d_in, const int* in_sizes, int n_in,
                              void* d_out, int out_size, void* d_ws, size_t ws_size,
                              hipStream_t stream) {
    (void)in_sizes; (void)n_in; (void)ws_size; (void)out_size;

    const float* x   = (const float*)d_in[0];
    const float* We1 = (const float*)d_in[1];  const float* be1 = (const float*)d_in[2];
    const float* We2 = (const float*)d_in[3];  const float* be2 = (const float*)d_in[4];
    const float* We3 = (const float*)d_in[5];  const float* be3 = (const float*)d_in[6];
    const float* We4 = (const float*)d_in[7];  const float* be4 = (const float*)d_in[8];
    const float* emb = (const float*)d_in[9];
    const float* Wd1 = (const float*)d_in[10]; const float* bd1 = (const float*)d_in[11];
    const float* Wd2 = (const float*)d_in[12]; const float* bd2 = (const float*)d_in[13];
    const float* Wd3 = (const float*)d_in[14]; const float* bd3 = (const float*)d_in[15];
    const float* Wd4 = (const float*)d_in[16]; const float* bd4 = (const float*)d_in[17];

    const size_t N_X  = (size_t)BATCH * 80 * 80 * 1;
    const size_t N_ZQ = (size_t)BATCH * 10 * 10 * LATENT;

    char* ws = (char*)d_ws;
    size_t off = 0;
    auto alloc16 = [&](size_t elems) -> _Float16* {
        off = (off + 255) & ~(size_t)255;
        _Float16* p = (_Float16*)(ws + off);
        off += elems * sizeof(_Float16);
        return p;
    };

    _Float16* xh   = alloc16(N_X);
    _Float16* We1h = alloc16(9 * 1 * 32);
    _Float16* We2h = alloc16(9 * 32 * 64);
    _Float16* We3h = alloc16(9 * 64 * 128);
    _Float16* We4h = alloc16(1 * 128 * LATENT);
    _Float16* Wd1h = alloc16(9 * LATENT * 128);
    _Float16* Wd2h = alloc16(9 * 128 * 64);
    _Float16* Wd3h = alloc16(9 * 64 * 32);
    _Float16* Wd4h = alloc16(9 * 32 * 1);
    _Float16* zqb  = alloc16(N_ZQ);
    _Float16* bufA = alloc16((size_t)BATCH * 80 * 80 * 32);  // max occupant: d3
    _Float16* bufB = alloc16((size_t)BATCH * 40 * 40 * 64);  // max occupant: d2

    auto cvt = [&](const float* src, _Float16* dst, size_t n) {
        cvt_f32_to_f16<<<dim3((unsigned)((n + WG - 1) / WG)), dim3(WG), 0, stream>>>(src, dst, (int)n);
    };
    cvt(x,   xh,   N_X);
    cvt(We1, We1h, 9 * 1 * 32);
    cvt(We2, We2h, 9 * 32 * 64);
    cvt(We3, We3h, 9 * 64 * 128);
    cvt(We4, We4h, 1 * 128 * LATENT);
    cvt(Wd1, Wd1h, 9 * LATENT * 128);
    cvt(Wd2, Wd2h, 9 * 128 * 64);
    cvt(Wd3, Wd3h, 9 * 64 * 32);
    cvt(Wd4, Wd4h, 9 * 32 * 1);

    // ---- encoder: 80 -> 40 -> 20 -> 10, SAME s2 (pad_lo = 0) ----
    launch_conv<80,80,  1, 40,40, 32, 3,3, 2,0, 0,1, 1,2, false,false>(xh,   We1h, be1, bufA, nullptr, stream);
    launch_conv<40,40, 32, 20,20, 64, 3,3, 2,0, 0,1, 2,2, false,false>(bufA, We2h, be2, bufB, nullptr, stream);
    launch_conv<20,20, 64, 10,10,128, 3,3, 2,0, 0,1, 2,2, false,false>(bufB, We3h, be3, bufA, nullptr, stream);
    // 1x1 conv: pure GEMM -> async DMA staging of A (global_load_async_to_lds_b128)
    launch_conv<10,10,128, 10,10, 16, 1,1, 1,0, 0,0, 1,1, false,true >(bufA, We4h, be4, bufB, nullptr, stream);

    // ---- vector quantize: z (bufB) -> zq ----
    {
        int N = BATCH * 10 * 10;
        vq_kernel<<<dim3((unsigned)((N + WG - 1) / WG)), dim3(WG), 0, stream>>>(bufB, emb, zqb, N);
    }

    // ---- decoder: 10 -> 20 -> 40 -> 80, fractionally-strided (pad = 1) ----
    launch_conv<10,10, 16, 20,20,128, 3,3, 2,1, 1,1, 2,2, false,false>(zqb,  Wd1h, bd1, bufA, nullptr, stream);
    launch_conv<20,20,128, 40,40, 64, 3,3, 2,1, 1,1, 2,2, false,false>(bufA, Wd2h, bd2, bufB, nullptr, stream);
    launch_conv<40,40, 64, 80,80, 32, 3,3, 2,1, 1,1, 1,2, false,false>(bufB, Wd3h, bd3, bufA, nullptr, stream);
    launch_conv<80,80, 32, 80,80,  1, 3,3, 1,1, 1,0, 1,1, true ,false>(bufA, Wd4h, bd4, nullptr, (float*)d_out, stream);
}
